// Bert_PF_50560355008538
// MI455X (gfx1250) — matrix operand order repaired and checked by reading, not compiled, not run on hardware
//
#include <hip/hip_runtime.h>
#include <hip/hip_bf16.h>
#include <stdint.h>
#include <stddef.h>

// ---------------------------------------------------------------------------
// BERT_PF forward for MI455X (gfx1250, wave32, WMMA).
//
// Precision strategy (roofline): dominant 120 GFLOP of GEMMs -> bf16 WMMA
// (v_wmma_f32_16x16x32_bf16) with f32 accumulate. Weights converted f32->bf16
// once per launch into OUTPUT-MAJOR [N][K] layout so the GEMM's B tile is a
// per-lane 16B async global->LDS copy (global_load_async_to_lds_b128,
// ASYNCcnt). A tiles are converted f32->bf16 via native casts and stored as
// packed b128. Attention (<1% FLOPs) and normalization/pooling are VALU.
//
// d_in layout assumption: depth-first, insertion-order flattening of
// setup_inputs(): 0..5 int inputs, then params leaves:
//  6 word_emb 7 pos_emb 8 type_emb 9 emb_g 10 emb_b
//  11 Wq 12 bq 13 Wk 14 bk 15 Wv 16 bv 17 Wo 18 bo 19 ln1_g 20 ln1_b
//  21 Wi 22 bi 23 Wo2 24 bo2 25 ln2_g 26 ln2_b 27 Wp 28 bp
//  29+gi*4+fi Wt[gi][fi]   45+gi*4+fi bt[gi][fi]   61+gi bconv[gi]
//  65 Wfc 66 bfc
// ---------------------------------------------------------------------------

#define BB   16
#define SS   128
#define STT  8
#define DD   768
#define LL   4
#define HH   12
#define DFF_ 3072
#define HDD  64
#define NFF  8

// ---------------- WMMA types ----------------
typedef __attribute__((ext_vector_type(16))) __bf16 v16bf;
typedef __attribute__((ext_vector_type(8)))  float  v8f;
typedef __attribute__((ext_vector_type(4)))  unsigned int u32x4;

union Frag16 { v16bf v; u32x4 q[2]; };
union PackBF { __bf16 h[16]; u32x4 q[2]; };

__device__ __forceinline__ unsigned int f2bf(float x) {
  unsigned int u = __float_as_uint(x);
  u += 0x7FFFu + ((u >> 16) & 1u);          // round-to-nearest-even
  return u >> 16;
}

// ---------------- f32 -> bf16, same layout (used for conv weights [C][K]) --
__global__ void cvt_bf16_kernel(const float* __restrict__ in,
                                unsigned short* __restrict__ out, size_t n) {
  for (size_t i = (size_t)blockIdx.x * blockDim.x + threadIdx.x; i < n;
       i += (size_t)gridDim.x * blockDim.x)
    out[i] = (unsigned short)f2bf(in[i]);
}

// f32 [L][K][N]  ->  bf16 [L][N][K]   (coalesced writes, strided reads)
__global__ void cvt_bf16_t_kernel(const float* __restrict__ in,
                                  unsigned short* __restrict__ out,
                                  int Lb, int Kc, int Nc) {
  size_t per = (size_t)Kc * Nc;
  size_t n = per * Lb;
  for (size_t i = (size_t)blockIdx.x * blockDim.x + threadIdx.x; i < n;
       i += (size_t)gridDim.x * blockDim.x) {
    size_t l = i / per, r = i % per;
    int nn = (int)(r / Kc), kk = (int)(r % Kc);
    out[i] = (unsigned short)f2bf(in[l * per + (size_t)kk * Nc + nn]);
  }
}

// ---------------- WMMA GEMM: C[M,N] = act(A[M,K](f32) * W[N,K](bf16)^T + b) -
// Workgroup: 256 threads = 8 waves; tile 128(M) x 64(N), K-step 32.
// Wave grid 4x2, each wave owns a 32x32 tile = 2x2 WMMA fragments.
#define BM 128
#define BN 64
#define BK 32
#define ASTR 40   // 32 + 8 pad -> 80B row stride (16B aligned)
#define BSTR 40

#define ACT_NONE 0
#define ACT_GELU 1
#define ACT_TANH 2
#define ACT_RELU 3

__global__ __launch_bounds__(256) void gemm_bf16_kernel(
    const float* __restrict__ A, const unsigned short* __restrict__ W,
    const float* __restrict__ bias, float* __restrict__ C,
    int M, int N, int K, int act) {
  __shared__ __align__(16) unsigned short As[BM * ASTR];  // [m][k]
  __shared__ __align__(16) unsigned short Bs[BN * BSTR];  // [n][k]

  const int tid  = threadIdx.x;
  const int lane = tid & 31;
  const int wave = tid >> 5;
  const int wm   = wave >> 1;  // 0..3 (M)
  const int wn   = wave & 1;   // 0..1 (N)
  const int row0 = blockIdx.y * BM;
  const int col0 = blockIdx.x * BN;

  const int hlf = lane >> 4;   // 0/1 half-wave
  const int l15 = lane & 15;

  v8f zero = {0.f, 0.f, 0.f, 0.f, 0.f, 0.f, 0.f, 0.f};
  v8f acc[2][2];
  acc[0][0] = zero; acc[0][1] = zero; acc[1][0] = zero; acc[1][1] = zero;

  // global->LDS mapping
  const int arow = tid >> 1;            // 0..127
  const int akb  = (tid & 1) * 16;      // 0/16
  const int bn   = tid >> 2;            // 0..63 (output column)
  const int bkb  = (tid & 3) * 8;       // 0,8,16,24 (k offset)

  // A row: clamp instead of branch; out-of-range rows duplicate row M-1 and
  // their results are dropped by the guarded store.
  int gr = row0 + arow;
  if (gr > M - 1) gr = M - 1;
  const float* ap = A + (size_t)gr * K + akb;

  // B tile LDS destination (32-bit LDS offset = low bits of flat address)
  unsigned ldsB = (unsigned)(size_t)&Bs[bn * BSTR + bkb];
  const unsigned short* wp = W + (size_t)(col0 + bn) * K + bkb;

  for (int k0 = 0; k0 < K; k0 += BK) {
    __syncthreads();
    {  // B tile: async global->LDS, 16B per lane, no VGPR round-trip
      unsigned long long ga = (unsigned long long)(wp + k0);
      asm volatile("global_load_async_to_lds_b128 %0, %1, off"
                   :: "v"(ldsB), "v"(ga) : "memory");
    }
    {  // A tile: 128x32 f32 -> bf16 (native converts), 2x b128 LDS stores
      PackBF pk;
#pragma unroll
      for (int i = 0; i < 16; ++i) pk.h[i] = (__bf16)ap[k0 + i];
      u32x4* dst = (u32x4*)&As[arow * ASTR + akb];
      dst[0] = pk.q[0];
      dst[1] = pk.q[1];
    }
    asm volatile("s_wait_asynccnt 0x0" ::: "memory");
    __syncthreads();

    // A fragments: 16x32 bf16 layout (ISA 7.12.2): lane half sets K base
    Frag16 fa[2], fb[2];
#pragma unroll
    for (int i = 0; i < 2; ++i) {
      const unsigned short* pa = &As[(wm * 32 + i * 16 + l15) * ASTR + hlf * 8];
      fa[i].q[0] = *(const u32x4*)pa;
      fa[i].q[1] = *(const u32x4*)(pa + 16);
    }
    // B fragments: 32x16 bf16; lane=col, half h holds K 8h..8h+7 and 16+8h..16+8h+7
#pragma unroll
    for (int j = 0; j < 2; ++j) {
      const unsigned short* pb = &Bs[(wn * 32 + j * 16 + l15) * BSTR + hlf * 8];
      fb[j].q[0] = *(const u32x4*)pb;
      fb[j].q[1] = *(const u32x4*)(pb + 16);
    }
#pragma unroll
    for (int i = 0; i < 2; ++i)
#pragma unroll
      for (int j = 0; j < 2; ++j)
        acc[i][j] = __builtin_amdgcn_wmma_f32_16x16x32_bf16(
            false, fa[i].v, false, fb[j].v, (short)0, acc[i][j], false, false);
  }

  // epilogue: C/D layout — VGPR r holds row r + 8*half, col = lane&15
#pragma unroll
  for (int i = 0; i < 2; ++i)
#pragma unroll
    for (int j = 0; j < 2; ++j) {
      int col = col0 + wn * 32 + j * 16 + l15;
      float bv = bias ? bias[col] : 0.0f;
#pragma unroll
      for (int r = 0; r < 8; ++r) {
        int m = row0 + wm * 32 + i * 16 + hlf * 8 + r;
        if (m < M) {
          float v = acc[i][j][r] + bv;
          if (act == ACT_GELU)      v = 0.5f * v * (1.0f + erff(v * 0.70710678118654752f));
          else if (act == ACT_TANH) v = tanhf(v);
          else if (act == ACT_RELU) v = fmaxf(v, 0.0f);
          C[(size_t)m * N + col] = v;
        }
      }
    }
}

// ---------------- embedding gather + LayerNorm (row per block) -------------
__global__ __launch_bounds__(256) void embed_ln_kernel(
    const int* __restrict__ ids, const int* __restrict__ tids,
    const float* __restrict__ we, const float* __restrict__ pe,
    const float* __restrict__ te, const float* __restrict__ g,
    const float* __restrict__ bv, float* __restrict__ out, int Sx) {
  __shared__ float red[256];
  int row = blockIdx.x;
  int s   = row % Sx;
  int id  = ids[row];
  int tt  = tids[row];
  int t   = threadIdx.x;
  float v[3]; float sum = 0.f;
#pragma unroll
  for (int i = 0; i < 3; ++i) {
    int c = t + i * 256;
    v[i] = we[(size_t)id * DD + c] + pe[(size_t)s * DD + c] + te[(size_t)tt * DD + c];
    sum += v[i];
  }
  red[t] = sum; __syncthreads();
  for (int o = 128; o > 0; o >>= 1) { if (t < o) red[t] += red[t + o]; __syncthreads(); }
  float mean = red[0] * (1.0f / DD); __syncthreads();
  sum = 0.f;
#pragma unroll
  for (int i = 0; i < 3; ++i) { float d = v[i] - mean; sum += d * d; }
  red[t] = sum; __syncthreads();
  for (int o = 128; o > 0; o >>= 1) { if (t < o) red[t] += red[t + o]; __syncthreads(); }
  float rstd = rsqrtf(red[0] * (1.0f / DD) + 1e-12f);
#pragma unroll
  for (int i = 0; i < 3; ++i) {
    int c = t + i * 256;
    out[(size_t)row * DD + c] = (v[i] - mean) * rstd * g[c] + bv[c];
  }
}

// ---------------- X = LN(X + R) (row per block) ----------------------------
__global__ __launch_bounds__(256) void ln_res_kernel(
    float* __restrict__ X, const float* __restrict__ R,
    const float* __restrict__ g, const float* __restrict__ bv) {
  __shared__ float red[256];
  int row = blockIdx.x;
  int t   = threadIdx.x;
  size_t base = (size_t)row * DD;
  float v[3]; float sum = 0.f;
#pragma unroll
  for (int i = 0; i < 3; ++i) {
    int c = t + i * 256;
    v[i] = X[base + c] + R[base + c];
    sum += v[i];
  }
  red[t] = sum; __syncthreads();
  for (int o = 128; o > 0; o >>= 1) { if (t < o) red[t] += red[t + o]; __syncthreads(); }
  float mean = red[0] * (1.0f / DD); __syncthreads();
  sum = 0.f;
#pragma unroll
  for (int i = 0; i < 3; ++i) { float d = v[i] - mean; sum += d * d; }
  red[t] = sum; __syncthreads();
  for (int o = 128; o > 0; o >>= 1) { if (t < o) red[t] += red[t + o]; __syncthreads(); }
  float rstd = rsqrtf(red[0] * (1.0f / DD) + 1e-12f);
#pragma unroll
  for (int i = 0; i < 3; ++i) {
    int c = t + i * 256;
    X[base + c] = (v[i] - mean) * rstd * g[c] + bv[c];
  }
}

// ---------------- attention: one (b,h,q) per block (128 thr) ---------------
__global__ __launch_bounds__(128) void attn_kernel(
    const float* __restrict__ Q, const float* __restrict__ Kx,
    const float* __restrict__ V, const int* __restrict__ mask,
    float* __restrict__ O, int Sx, float scale) {
  __shared__ float qs[HDD];
  __shared__ float sc[128];
  __shared__ float red[128];
  int blk = blockIdx.x;
  int q  = blk % Sx;
  int bh = blk / Sx;
  int h  = bh % HH;
  int b  = bh / HH;
  int t  = threadIdx.x;
  size_t rowQ = ((size_t)(b * Sx + q)) * DD + h * HDD;
  if (t < HDD) qs[t] = Q[rowQ + t];
  __syncthreads();
  float s = -1e30f;
  if (t < Sx) {
    const float* kp = Kx + ((size_t)(b * Sx + t)) * DD + h * HDD;
    float d = 0.f;
#pragma unroll 8
    for (int i = 0; i < HDD; ++i) d += qs[i] * kp[i];
    s = d * scale + (mask[b * Sx + t] ? 0.0f : -10000.0f);
  }
  sc[t] = s; red[t] = s; __syncthreads();
  for (int o = 64; o > 0; o >>= 1) { if (t < o) red[t] = fmaxf(red[t], red[t + o]); __syncthreads(); }
  float mx = red[0]; __syncthreads();
  float e = (t < Sx) ? __expf(sc[t] - mx) : 0.f;
  sc[t] = e; red[t] = e; __syncthreads();
  for (int o = 64; o > 0; o >>= 1) { if (t < o) red[t] += red[t + o]; __syncthreads(); }
  float inv = 1.0f / red[0];
  if (t < HDD) {
    float o = 0.f;
    for (int k = 0; k < Sx; ++k)
      o += sc[k] * V[((size_t)(b * Sx + k)) * DD + h * HDD + t];
    O[rowQ + t] = o * inv;
  }
}

// ---------------- gather h[:,0,:] rows -------------------------------------
__global__ void row0_kernel(const float* __restrict__ Hs, float* __restrict__ out) {
  int idx = blockIdx.x * blockDim.x + threadIdx.x;
  if (idx < BB * DD) {
    int b = idx / DD, c = idx % DD;
    out[idx] = Hs[((size_t)b * SS) * DD + c];
  }
}

// ---------------- im2col windows for target branch -------------------------
__global__ void win_kernel(const float* __restrict__ Hsrc, float* __restrict__ win,
                           int T, int fs, int Ssrc, size_t total) {
  for (size_t idx = (size_t)blockIdx.x * blockDim.x + threadIdx.x; idx < total;
       idx += (size_t)gridDim.x * blockDim.x) {
    int d  = (int)(idx % DD);
    size_t r = idx / DD;
    int kk = (int)(r % fs); r /= fs;
    int t  = (int)(r % T);
    int b  = (int)(r / T);
    win[idx] = Hsrc[((size_t)(b * Ssrc + t + kk)) * DD + d];
  }
}

// ---------------- mean over time into cat_g slice --------------------------
__global__ void meanpool_kernel(const float* __restrict__ CO, float* __restrict__ catg,
                                int T, int Cg, int colOff, int catW, size_t total) {
  for (size_t idx = (size_t)blockIdx.x * blockDim.x + threadIdx.x; idx < total;
       idx += (size_t)gridDim.x * blockDim.x) {
    int b = (int)(idx / Cg), c = (int)(idx % Cg);
    float s = 0.f;
    for (int t = 0; t < T; ++t) s += CO[((size_t)(b * T + t)) * Cg + c];
    catg[(size_t)b * catW + colOff + c] = s * (1.0f / T);
  }
}

// ---------------- dynamic per-sample conv + relu + max-pool ----------------
__global__ __launch_bounds__(256) void dynconv_kernel(
    const float* __restrict__ Hs, const float* __restrict__ catg,
    const float* __restrict__ bconv, float* __restrict__ pparts,
    int g, int T, int gi, int Sx) {
  __shared__ float red[256];
  int b = blockIdx.x >> 3;   // NF = 8
  int f = blockIdx.x & 7;
  int gD = g * DD;
  const float* fil = catg + (size_t)b * (NFF * gD) + (size_t)f * gD;
  float vmax = -1e30f;
  for (int t = 0; t < T; ++t) {
    float p = 0.f;
    for (int j = threadIdx.x; j < gD; j += 256) {
      int kk = j / DD, d = j - kk * DD;
      p += Hs[((size_t)(b * Sx + t + kk)) * DD + d] * fil[j];
    }
    red[threadIdx.x] = p; __syncthreads();
    for (int o = 128; o > 0; o >>= 1) { if (threadIdx.x < o) red[threadIdx.x] += red[threadIdx.x + o]; __syncthreads(); }
    float val = fmaxf(red[0] + bconv[f], 0.0f);
    __syncthreads();
    vmax = fmaxf(vmax, val);
  }
  if (threadIdx.x == 0) pparts[b * 32 + gi * NFF + f] = vmax;
}

// ---------------- final FC + tanh ------------------------------------------
__global__ void fc_out_kernel(const float* __restrict__ pooled,
                              const float* __restrict__ pparts,
                              const float* __restrict__ Wfc,
                              const float* __restrict__ bfc,
                              float* __restrict__ out) {
  int idx = threadIdx.x;
  if (idx >= BB * 3) return;
  int b = idx / 3, o = idx % 3;
  float s = bfc[o];
  for (int j = 0; j < DD; ++j) s += pooled[(size_t)b * DD + j] * Wfc[j * 3 + o];
  for (int j = 0; j < 32; ++j) s += pparts[b * 32 + j] * Wfc[(DD + j) * 3 + o];
  out[b * 3 + o] = tanhf(s);
}

// ---------------------------------------------------------------------------
static inline void gemm(const float* A, const unsigned short* W, const float* bias,
                        float* C, int M, int N, int K, int act, hipStream_t s) {
  dim3 grid((unsigned)(N / BN), (unsigned)((M + BM - 1) / BM));
  gemm_bf16_kernel<<<grid, 256, 0, s>>>(A, W, bias, C, M, N, K, act);
}

extern "C" void kernel_launch(void* const* d_in, const int* in_sizes, int n_in,
                              void* d_out, int out_size, void* d_ws, size_t ws_size,
                              hipStream_t stream) {
  (void)in_sizes; (void)n_in; (void)out_size; (void)ws_size;
  const int* input_ids      = (const int*)d_in[0];
  const int* token_type_ids = (const int*)d_in[1];
  const int* attention_mask = (const int*)d_in[2];
  const int* input_t_ids    = (const int*)d_in[3];
  const int* input_t_mask   = (const int*)d_in[4];
  const int* segment_t_ids  = (const int*)d_in[5];
  const float* word_emb = (const float*)d_in[6];
  const float* pos_emb  = (const float*)d_in[7];
  const float* type_emb = (const float*)d_in[8];
  const float* emb_g    = (const float*)d_in[9];
  const float* emb_b    = (const float*)d_in[10];
  const float* Wq  = (const float*)d_in[11]; const float* bq  = (const float*)d_in[12];
  const float* Wk  = (const float*)d_in[13]; const float* bk  = (const float*)d_in[14];
  const float* Wv  = (const float*)d_in[15]; const float* bvv = (const float*)d_in[16];
  const float* Wo  = (const float*)d_in[17]; const float* bo  = (const float*)d_in[18];
  const float* ln1g = (const float*)d_in[19]; const float* ln1b = (const float*)d_in[20];
  const float* Wi  = (const float*)d_in[21]; const float* bi  = (const float*)d_in[22];
  const float* Wo2 = (const float*)d_in[23]; const float* bo2 = (const float*)d_in[24];
  const float* ln2g = (const float*)d_in[25]; const float* ln2b = (const float*)d_in[26];
  const float* Wp = (const float*)d_in[27]; const float* bp = (const float*)d_in[28];
  const float* Wfc = (const float*)d_in[65]; const float* bfc = (const float*)d_in[66];

  // -------- workspace carve-out --------
  char* ws = (char*)d_ws;
  size_t off = 0;
  auto take = [&](size_t bytes) -> char* {
    char* p = ws + off;
    off = (off + bytes + 255) & ~(size_t)255;
    return p;
  };
  const size_t nDD  = (size_t)LL * DD * DD;
  const size_t nDF  = (size_t)LL * DD * DFF_;
  unsigned short* wq_bf  = (unsigned short*)take(nDD * 2);   // [L][N][K]
  unsigned short* wk_bf  = (unsigned short*)take(nDD * 2);
  unsigned short* wv_bf  = (unsigned short*)take(nDD * 2);
  unsigned short* wo_bf  = (unsigned short*)take(nDD * 2);
  unsigned short* wi_bf  = (unsigned short*)take(nDF * 2);   // [L][DFF][D]
  unsigned short* wo2_bf = (unsigned short*)take(nDF * 2);   // [L][D][DFF]
  unsigned short* wp_bf  = (unsigned short*)take((size_t)DD * DD * 2);
  unsigned short* wt_bf  = (unsigned short*)take((size_t)6144 * 3072 * 2); // reused [C][K] slot
  float* Hs  = (float*)take((size_t)BB * SS * DD * 4);
  float* Ht  = (float*)take((size_t)BB * STT * DD * 4);
  float* Qb  = (float*)take((size_t)BB * SS * DD * 4);
  float* Kb  = (float*)take((size_t)BB * SS * DD * 4);
  float* Vb  = (float*)take((size_t)BB * SS * DD * 4);
  float* T1  = (float*)take((size_t)BB * SS * DD * 4);
  float* T2  = (float*)take((size_t)BB * SS * DD * 4);
  float* FF  = (float*)take((size_t)BB * SS * DFF_ * 4);
  float* WIN = (float*)take((size_t)128 * 3072 * 4);
  float* CO  = (float*)take((size_t)128 * 6144 * 4);
  float* CATG = (float*)take((size_t)BB * 24576 * 4);
  float* POOLH  = (float*)take((size_t)BB * DD * 4);
  float* POOLED = (float*)take((size_t)BB * DD * 4);
  float* PPARTS = (float*)take((size_t)BB * 32 * 4);

  // -------- convert shared weights to bf16 [N][K] once --------
  cvt_bf16_t_kernel<<<2048, 256, 0, stream>>>(Wq,  wq_bf,  LL, DD,   DD);
  cvt_bf16_t_kernel<<<2048, 256, 0, stream>>>(Wk,  wk_bf,  LL, DD,   DD);
  cvt_bf16_t_kernel<<<2048, 256, 0, stream>>>(Wv,  wv_bf,  LL, DD,   DD);
  cvt_bf16_t_kernel<<<2048, 256, 0, stream>>>(Wo,  wo_bf,  LL, DD,   DD);
  cvt_bf16_t_kernel<<<2048, 256, 0, stream>>>(Wi,  wi_bf,  LL, DD,   DFF_);
  cvt_bf16_t_kernel<<<2048, 256, 0, stream>>>(Wo2, wo2_bf, LL, DFF_, DD);
  cvt_bf16_t_kernel<<<2048, 256, 0, stream>>>(Wp,  wp_bf,  1,  DD,   DD);

  // -------- embeddings + LN --------
  embed_ln_kernel<<<BB * SS, 256, 0, stream>>>(input_ids, token_type_ids, word_emb,
                                               pos_emb, type_emb, emb_g, emb_b, Hs, SS);
  embed_ln_kernel<<<BB * STT, 256, 0, stream>>>(input_t_ids, segment_t_ids, word_emb,
                                                pos_emb, type_emb, emb_g, emb_b, Ht, STT);

  // -------- encoder (shared weights, run twice) --------
  auto encode = [&](float* Hbuf, int Sx, const int* mask) {
    int M = BB * Sx;
    for (int l = 0; l < LL; ++l) {
      const size_t odd = (size_t)l * DD * DD;
      const size_t odf = (size_t)l * DD * DFF_;
      gemm(Hbuf, wq_bf + odd, bq  + l * DD, Qb, M, DD, DD, ACT_NONE, stream);
      gemm(Hbuf, wk_bf + odd, bk  + l * DD, Kb, M, DD, DD, ACT_NONE, stream);
      gemm(Hbuf, wv_bf + odd, bvv + l * DD, Vb, M, DD, DD, ACT_NONE, stream);
      attn_kernel<<<BB * HH * Sx, 128, 0, stream>>>(Qb, Kb, Vb, mask, T1, Sx, 0.125f);
      gemm(T1, wo_bf + odd, bo + l * DD, T2, M, DD, DD, ACT_NONE, stream);
      ln_res_kernel<<<M, 256, 0, stream>>>(Hbuf, T2, ln1g + l * DD, ln1b + l * DD);
      gemm(Hbuf, wi_bf + odf, bi + l * DFF_, FF, M, DFF_, DD, ACT_GELU, stream);
      gemm(FF, wo2_bf + odf, bo2 + l * DD, T2, M, DD, DFF_, ACT_NONE, stream);
      ln_res_kernel<<<M, 256, 0, stream>>>(Hbuf, T2, ln2g + l * DD, ln2b + l * DD);
    }
  };
  encode(Hs, SS, attention_mask);
  encode(Ht, STT, input_t_mask);

  // -------- pooler: tanh(h[:,0] @ Wp + bp) --------
  row0_kernel<<<(BB * DD + 255) / 256, 256, 0, stream>>>(Hs, POOLH);
  gemm(POOLH, wp_bf, bp, POOLED, BB, DD, DD, ACT_TANH, stream);

  // -------- target branch: static convs -> dynamic filters -> dynamic conv --
  for (int gi = 0; gi < 4; ++gi) {
    int g    = gi + 1;
    int Cg   = g * NFF * DD / 4;  // g*1536
    int catW = 4 * Cg;            // == NF * g * D
    for (int fi = 0; fi < 4; ++fi) {
      int fs = fi + 1;
      int T  = STT - fs + 1;
      int M  = BB * T;
      int K  = fs * DD;
      size_t nwin = (size_t)M * K;
      win_kernel<<<(unsigned)((nwin + 255) / 256), 256, 0, stream>>>(Ht, WIN, T, fs, STT, nwin);
      // Wt[gi][fi] is (C, fs, D) == [C][K]: already output-major, plain convert
      cvt_bf16_kernel<<<2048, 256, 0, stream>>>((const float*)d_in[29 + gi * 4 + fi],
                                                wt_bf, (size_t)Cg * K);
      gemm(WIN, wt_bf, (const float*)d_in[45 + gi * 4 + fi], CO, M, Cg, K, ACT_RELU, stream);
      size_t npool = (size_t)BB * Cg;
      meanpool_kernel<<<(unsigned)((npool + 255) / 256), 256, 0, stream>>>(
          CO, CATG, T, Cg, fi * Cg, catW, npool);
    }
    int Tm = SS - g + 1;
    dynconv_kernel<<<BB * NFF, 256, 0, stream>>>(Hs, CATG, (const float*)d_in[61 + gi],
                                                 PPARTS, g, Tm, gi, SS);
  }

  // -------- final FC --------
  fc_out_kernel<<<1, 64, 0, stream>>>(POOLED, PPARTS, Wfc, bfc, (float*)d_out);
}
